// DiffAttn_87273735455387
// MI455X (gfx1250) — matrix operand order repaired and checked by reading, not compiled, not run on hardware
//
#include <hip/hip_runtime.h>
#include <cstdint>
#include <cstddef>

// ---------- CDNA5 WMMA types ----------
typedef __bf16 bf16;
typedef __attribute__((ext_vector_type(16))) __bf16 v16bf;
typedef __attribute__((ext_vector_type(8)))  float  v8f;

#define T_SEQ 2048
#define D_DIM 2048
#define NUM_H 16
#define HD    64            // half head dim (q/k sub-head dim)
#define LAMBDA_INIT_F 0.783605766f
#define ONE_MINUS_LI  0.216394234f
#define LN_EPS_F      1e-5f

// ---- CDNA5 async global->LDS copy (ASYNCcnt path) ----
#if defined(__has_builtin)
#  if __has_builtin(__builtin_amdgcn_s_wait_asynccnt)
#    define WAIT_ASYNC(n) __builtin_amdgcn_s_wait_asynccnt(n)
#  endif
#endif
#ifndef WAIT_ASYNC
#  define WAIT_ASYNC(n) asm volatile("s_wait_asynccnt %0" :: "n"(n) : "memory")
#endif

__device__ __forceinline__ void async_copy_b128(void* lds_dst, const void* gsrc) {
    // VDST = LDS byte address (generic shared addr truncates to LDS offset,
    // ISA 10.2 aperture rule), VADDR = 64-bit global address, GV mode.
    uint32_t l = (uint32_t)(size_t)lds_dst;
    uint64_t g = (uint64_t)(size_t)gsrc;
    asm volatile("global_load_async_to_lds_b128 %0, %1, off"
                 :: "v"(l), "v"(g) : "memory");
}

__device__ __forceinline__ bf16 f2bf(float f) {
    union { float f; uint32_t u; } x; x.f = f;
    uint32_t r = x.u + 0x7FFFu + ((x.u >> 16) & 1u);   // RNE
    union { uint16_t s; bf16 b; } y; y.s = (uint16_t)(r >> 16);
    return y.b;
}

// ---------------------------------------------------------------------------
// Cast fp32 -> bf16 (optionally scaled; q-scale hd^-0.5 folded into Wq)
// ---------------------------------------------------------------------------
__global__ void cast_bf16_kernel(const float* __restrict__ in,
                                 bf16* __restrict__ out, int n, float scale) {
    for (int i = blockIdx.x * blockDim.x + threadIdx.x; i < n;
         i += gridDim.x * blockDim.x)
        out[i] = f2bf(in[i] * scale);
}

// ---------------------------------------------------------------------------
// lambda = exp(sum(lq1*lk1)) - exp(sum(lq2*lk2)) + LAMBDA_INIT   (hd=64)
// ---------------------------------------------------------------------------
__global__ void lambda_kernel(const float* lq1, const float* lk1,
                              const float* lq2, const float* lk2, float* lam) {
    int j = threadIdx.x;                 // 64 threads = 2 waves
    float a = lq1[j] * lk1[j];
    float b = lq2[j] * lk2[j];
    #pragma unroll
    for (int off = 16; off > 0; off >>= 1) {
        a += __shfl_xor(a, off, 32);
        b += __shfl_xor(b, off, 32);
    }
    __shared__ float sa[2], sb[2];
    if ((j & 31) == 0) { sa[j >> 5] = a; sb[j >> 5] = b; }
    __syncthreads();
    if (j == 0)
        *lam = __expf(sa[0] + sa[1]) - __expf(sb[0] + sb[1]) + LAMBDA_INIT_F;
}

// ---------------------------------------------------------------------------
// Tiled bf16 WMMA GEMM: C(MxN) = A(MxK) * B(KxN), row-major everywhere.
// Block = 256 threads = 8 waves; tile 128x128x64; double-buffered LDS filled
// by global_load_async_to_lds_b128 (ASYNCcnt), 16 WMMAs per wave per stage.
// ---------------------------------------------------------------------------
template <bool F32OUT>
__global__ __launch_bounds__(256)
void gemm_wmma_kernel(const bf16* __restrict__ A, const bf16* __restrict__ B,
                      void* __restrict__ Cout, int M, int N, int K) {
    __shared__ bf16 As[2][128][64];     // 2 x 16 KB
    __shared__ bf16 Bs[2][64][128];     // 2 x 16 KB

    const int tid  = threadIdx.x;
    const int lane = tid & 31;
    const int wave = tid >> 5;
    const int nlane = lane & 15, half = lane >> 4;
    const int m0 = blockIdx.y * 128, n0 = blockIdx.x * 128;
    const int wrow = wave * 16;
    const int kbA = half * 8;      // A-frag K base (ISA 16-bit A 16x32 layout)
    const int kbB = half * 8;      // B-frag K base (same table order as A)

    v8f acc[8];
    #pragma unroll
    for (int i = 0; i < 8; i++) acc[i] = (v8f){0,0,0,0,0,0,0,0};

    // issue one 32 KB stage (8 async b128 per thread = 8 per wave)
    auto issue = [&](int k0, int b) {
        #pragma unroll
        for (int c = tid; c < 1024; c += 256) {         // A: 128 x 64 bf16
            int r = c >> 3, col = (c & 7) * 8;
            async_copy_b128(&As[b][r][col],
                            &A[(size_t)(m0 + r) * K + k0 + col]);
        }
        #pragma unroll
        for (int c = tid; c < 1024; c += 256) {         // B: 64 x 128 bf16
            int r = c >> 4, col = (c & 15) * 8;
            async_copy_b128(&Bs[b][r][col],
                            &B[(size_t)(k0 + r) * N + n0 + col]);
        }
    };

    int buf = 0;
    issue(0, 0);
    for (int k0 = 0; k0 < K; k0 += 64) {
        if (k0 + 64 < K) { issue(k0 + 64, buf ^ 1); WAIT_ASYNC(8); }
        else             { WAIT_ASYNC(0); }
        __syncthreads();                 // stage `buf` resident for all waves

        #pragma unroll
        for (int ks = 0; ks < 2; ks++) { // two K=32 sub-steps
            v16bf af;
            #pragma unroll
            for (int i = 0; i < 8; i++)
                af[i]     = As[buf][wrow + nlane][ks * 32 + kbA + i];
            #pragma unroll
            for (int i = 0; i < 8; i++)
                af[8 + i] = As[buf][wrow + nlane][ks * 32 + 16 + kbA + i];

            #pragma unroll
            for (int nt = 0; nt < 8; nt++) {
                v16bf bfr;
                #pragma unroll
                for (int i = 0; i < 16; i++)
                    bfr[i] = Bs[buf][ks * 32 + kbB + (i & 7) + ((i >> 3) << 4)][nt * 16 + nlane];
                acc[nt] = __builtin_amdgcn_wmma_f32_16x16x32_bf16(
                    false, af, false, bfr, (short)0, acc[nt], false, false);
            }
        }
        __syncthreads();                 // readers done before next overwrite
        buf ^= 1;
    }

    // ---- store (C layout: row = wrow + r + half*8, col = nt*16 + nlane) ----
    const int rh = half * 8;
    #pragma unroll
    for (int nt = 0; nt < 8; nt++) {
        #pragma unroll
        for (int r = 0; r < 8; r++) {
            size_t idx = (size_t)(m0 + wrow + rh + r) * N + n0 + nt * 16 + nlane;
            if (F32OUT) ((float*)Cout)[idx] = acc[nt][r];
            else        ((bf16*)Cout)[idx]  = f2bf(acc[nt][r]);
        }
    }
}

// ---------------------------------------------------------------------------
// Flash-style causal attention, one (sub-head s, 128-row block) per CTA.
// Q/K: (T, 32, 64) as (T,D) bf16; V: (T, 16, 128) as (T,D) bf16.
// O: fp32 [32][T][128].  Q already scaled by hd^-0.5 (folded into Wq).
// K/V tiles staged with async global->LDS DMA.
// ---------------------------------------------------------------------------
__global__ __launch_bounds__(256)
void flash_attn_kernel(const bf16* __restrict__ Qb, const bf16* __restrict__ Kb,
                       const bf16* __restrict__ Vb, float* __restrict__ O) {
    __shared__ bf16 Qs[128][64];
    __shared__ bf16 Ks[32][64];
    __shared__ bf16 Vs[32][128];
    __shared__ bf16 Ps[8][16][32];     // per-wave P staging

    const int s  = blockIdx.y;          // sub-head 0..31
    const int h  = s >> 1;              // v-head
    const int i0 = blockIdx.x * 128;
    const int tid = threadIdx.x;
    const int lane = tid & 31, wave = tid >> 5;
    const int nlane = lane & 15, half = lane >> 4;
    const int wrow = wave * 16;
    const int kbA = half * 8;

    // stage Q block (128 x 64) once
    #pragma unroll
    for (int t = tid; t < 128 * 32; t += 256) {
        int r = t >> 5, c = (t & 31) * 2;
        *(uint32_t*)&Qs[r][c] =
            *(const uint32_t*)&Qb[(size_t)(i0 + r) * D_DIM + s * HD + c];
    }
    __syncthreads();

    v8f acc[8];
    #pragma unroll
    for (int i = 0; i < 8; i++) acc[i] = (v8f){0,0,0,0,0,0,0,0};
    float m_run[8], l_run[8];
    #pragma unroll
    for (int r = 0; r < 8; r++) { m_run[r] = -1e30f; l_run[r] = 0.0f; }

    const int kend = i0 + 128;          // causal bound for this row block
    for (int kb = 0; kb < kend; kb += 32) {
        // ---- async stage K (32x64) and V (32x128) ----
        {
            int r = tid >> 3, c = (tid & 7) * 8;              // 256 chunks
            async_copy_b128(&Ks[r][c],
                            &Kb[(size_t)(kb + r) * D_DIM + s * HD + c]);
        }
        #pragma unroll
        for (int t = tid; t < 512; t += 256) {                // 512 chunks
            int r = t >> 4, c = (t & 15) * 8;
            async_copy_b128(&Vs[r][c],
                            &Vb[(size_t)(kb + r) * D_DIM + h * 128 + c]);
        }
        WAIT_ASYNC(0);
        __syncthreads();

        // ---- S tile (16 rows x 32 keys) = Q Kt, contraction hd=64 = 2 WMMAs ----
        v8f s0 = (v8f){0,0,0,0,0,0,0,0}, s1 = s0;
        #pragma unroll
        for (int ks = 0; ks < 2; ks++) {
            v16bf aq;
            #pragma unroll
            for (int i = 0; i < 8; i++) aq[i]     = Qs[wrow + nlane][ks * 32 + kbA + i];
            #pragma unroll
            for (int i = 0; i < 8; i++) aq[8 + i] = Qs[wrow + nlane][ks * 32 + 16 + kbA + i];
            v16bf b0, b1;
            #pragma unroll
            for (int i = 0; i < 16; i++) b0[i] = Ks[nlane][ks * 32 + half * 8 + (i & 7) + ((i >> 3) << 4)];
            #pragma unroll
            for (int i = 0; i < 16; i++) b1[i] = Ks[16 + nlane][ks * 32 + half * 8 + (i & 7) + ((i >> 3) << 4)];
            s0 = __builtin_amdgcn_wmma_f32_16x16x32_bf16(false, aq, false, b0, (short)0, s0, false, false);
            s1 = __builtin_amdgcn_wmma_f32_16x16x32_bf16(false, aq, false, b1, (short)0, s1, false, false);
        }

        // ---- causal mask + online softmax (per row; 16-lane half reductions) ----
        #pragma unroll
        for (int r = 0; r < 8; r++) {
            const int row_g = i0 + wrow + half * 8 + r;
            const int k0g = kb + nlane, k1g = kb + 16 + nlane;
            float v0 = (k0g <= row_g) ? s0[r] : -1e30f;
            float v1 = (k1g <= row_g) ? s1[r] : -1e30f;
            float mx = fmaxf(v0, v1);
            #pragma unroll
            for (int off = 1; off < 16; off <<= 1) mx = fmaxf(mx, __shfl_xor(mx, off, 16));
            float mnew = fmaxf(m_run[r], mx);
            float p0 = __expf(v0 - mnew), p1 = __expf(v1 - mnew);
            float ps = p0 + p1;
            #pragma unroll
            for (int off = 1; off < 16; off <<= 1) ps += __shfl_xor(ps, off, 16);
            float alpha = __expf(m_run[r] - mnew);
            l_run[r] = l_run[r] * alpha + ps;
            m_run[r] = mnew;
            #pragma unroll
            for (int nt = 0; nt < 8; nt++) acc[nt][r] *= alpha;
            Ps[wave][half * 8 + r][nlane]      = f2bf(p0);
            Ps[wave][half * 8 + r][16 + nlane] = f2bf(p1);
        }
        __syncthreads();

        // ---- O += P (16x32) * V (32x128) : 8 WMMAs ----
        v16bf ap;
        #pragma unroll
        for (int i = 0; i < 8; i++) ap[i]     = Ps[wave][nlane][kbA + i];
        #pragma unroll
        for (int i = 0; i < 8; i++) ap[8 + i] = Ps[wave][nlane][16 + kbA + i];
        #pragma unroll
        for (int nt = 0; nt < 8; nt++) {
            v16bf bv;
            #pragma unroll
            for (int i = 0; i < 16; i++) bv[i] = Vs[half * 8 + (i & 7) + ((i >> 3) << 4)][nt * 16 + nlane];
            acc[nt] = __builtin_amdgcn_wmma_f32_16x16x32_bf16(
                false, ap, false, bv, (short)0, acc[nt], false, false);
        }
        __syncthreads();   // Ks/Vs/Ps consumers done before next async stage
    }

    // ---- epilogue: normalize rows, store fp32 O[s][row][0..128) ----
    #pragma unroll
    for (int r = 0; r < 8; r++) {
        float inv = 1.0f / l_run[r];
        size_t base = ((size_t)s * T_SEQ + (i0 + wrow + half * 8 + r)) * 128;
        #pragma unroll
        for (int nt = 0; nt < 8; nt++)
            O[base + nt * 16 + nlane] = acc[nt][r] * inv;
    }
}

// ---------------------------------------------------------------------------
// diff = O0 - lam*O1, LayerNorm(128, no affine, biased var), *(1-lam_init),
// store bf16 flat in (H, T, 128) order == reference's no-transpose reshape.
// grid = H*T blocks of 128 threads.
// ---------------------------------------------------------------------------
__global__ __launch_bounds__(128)
void combine_ln_kernel(const float* __restrict__ O, const float* __restrict__ lam_p,
                       bf16* __restrict__ Zb) {
    const int idx = blockIdx.x;
    const int h = idx >> 11;            // / T_SEQ
    const int t = idx & (T_SEQ - 1);
    const int j = threadIdx.x;
    const float lam = *lam_p;

    float o0 = O[((size_t)(2 * h)     * T_SEQ + t) * 128 + j];
    float o1 = O[((size_t)(2 * h + 1) * T_SEQ + t) * 128 + j];
    float d = o0 - lam * o1;

    __shared__ float red[4];
    float ssum = d;
    #pragma unroll
    for (int off = 16; off > 0; off >>= 1) ssum += __shfl_xor(ssum, off, 32);
    if ((j & 31) == 0) red[j >> 5] = ssum;
    __syncthreads();
    float mean = (red[0] + red[1] + red[2] + red[3]) * (1.0f / 128.0f);

    float c = d - mean;
    float vs = c * c;
    #pragma unroll
    for (int off = 16; off > 0; off >>= 1) vs += __shfl_xor(vs, off, 32);
    __syncthreads();
    if ((j & 31) == 0) red[j >> 5] = vs;
    __syncthreads();
    float var = (red[0] + red[1] + red[2] + red[3]) * (1.0f / 128.0f);

    float y = c * rsqrtf(var + LN_EPS_F) * ONE_MINUS_LI;
    Zb[(size_t)h * T_SEQ * 128 + (size_t)t * 128 + j] = f2bf(y);
}

// ---------------------------------------------------------------------------
extern "C" void kernel_launch(void* const* d_in, const int* in_sizes, int n_in,
                              void* d_out, int out_size, void* d_ws, size_t ws_size,
                              hipStream_t stream) {
    const float* x   = (const float*)d_in[0];
    const float* Wq  = (const float*)d_in[1];
    const float* Wk  = (const float*)d_in[2];
    const float* Wv  = (const float*)d_in[3];
    const float* Wo  = (const float*)d_in[4];
    const float* lq1 = (const float*)d_in[5];
    const float* lk1 = (const float*)d_in[6];
    const float* lq2 = (const float*)d_in[7];
    const float* lk2 = (const float*)d_in[8];
    float* out = (float*)d_out;

    const int NE = T_SEQ * D_DIM;                       // 4M elements
    const size_t SZ_BF = (size_t)NE * sizeof(uint16_t); // 8 MiB per bf16 matrix
    char* ws = (char*)d_ws;
    bf16* Xb  = (bf16*)(ws + 0 * SZ_BF);
    bf16* Wqb = (bf16*)(ws + 1 * SZ_BF);
    bf16* Wkb = (bf16*)(ws + 2 * SZ_BF);
    bf16* Wvb = (bf16*)(ws + 3 * SZ_BF);
    bf16* Wob = (bf16*)(ws + 4 * SZ_BF);
    bf16* Qb  = (bf16*)(ws + 5 * SZ_BF);
    bf16* Kb  = (bf16*)(ws + 6 * SZ_BF);
    bf16* Vb  = (bf16*)(ws + 7 * SZ_BF);
    bf16* Zb  = (bf16*)(ws + 8 * SZ_BF);
    float* Obuf = (float*)(ws + 9 * SZ_BF);             // 32*2048*128 f32 = 32 MiB
    float* lam  = (float*)(ws + 9 * SZ_BF + (size_t)32 * T_SEQ * 128 * 4);

    const float qscale = 0.125f;                         // hd^-0.5, folded into Wq

    // 1) casts
    cast_bf16_kernel<<<1024, 256, 0, stream>>>(x,  Xb,  NE, 1.0f);
    cast_bf16_kernel<<<1024, 256, 0, stream>>>(Wq, Wqb, NE, qscale);
    cast_bf16_kernel<<<1024, 256, 0, stream>>>(Wk, Wkb, NE, 1.0f);
    cast_bf16_kernel<<<1024, 256, 0, stream>>>(Wv, Wvb, NE, 1.0f);
    cast_bf16_kernel<<<1024, 256, 0, stream>>>(Wo, Wob, NE, 1.0f);

    // 2) lambda scalar
    lambda_kernel<<<1, 64, 0, stream>>>(lq1, lk1, lq2, lk2, lam);

    // 3) projections: Q = X*(Wq*scale), K = X*Wk, V = X*Wv
    dim3 ggrid(D_DIM / 128, T_SEQ / 128);
    gemm_wmma_kernel<false><<<ggrid, 256, 0, stream>>>(Xb, Wqb, Qb, T_SEQ, D_DIM, D_DIM);
    gemm_wmma_kernel<false><<<ggrid, 256, 0, stream>>>(Xb, Wkb, Kb, T_SEQ, D_DIM, D_DIM);
    gemm_wmma_kernel<false><<<ggrid, 256, 0, stream>>>(Xb, Wvb, Vb, T_SEQ, D_DIM, D_DIM);

    // 4) dual-stream causal flash attention (32 sub-heads)
    dim3 fgrid(T_SEQ / 128, 32);
    flash_attn_kernel<<<fgrid, 256, 0, stream>>>(Qb, Kb, Vb, Obuf);

    // 5) diff + sub-LN + scale, written in faithful (H,T,128) flat order
    combine_ln_kernel<<<NUM_H * T_SEQ, 128, 0, stream>>>(Obuf, lam, Zb);

    // 6) final projection: out = Z * Wo (fp32 result)
    gemm_wmma_kernel<true><<<ggrid, 256, 0, stream>>>(Zb, Wob, out, T_SEQ, D_DIM, D_DIM);
}
